// Adp_4681514353398
// MI455X (gfx1250) — matrix operand order repaired and checked by reading, not compiled, not run on hardware
//
#include <hip/hip_runtime.h>

// ---------------------------------------------------------------------------
// Types for CDNA5 WMMA / TDM (gfx1250, wave32)
// ---------------------------------------------------------------------------
typedef __bf16 bf16_t;
typedef bf16_t v16bf __attribute__((ext_vector_type(16)));
typedef float  v8f   __attribute__((ext_vector_type(8)));
typedef unsigned int u32x4 __attribute__((ext_vector_type(4)));
typedef int          i32x4 __attribute__((ext_vector_type(4)));
typedef int          i32x8 __attribute__((ext_vector_type(8)));

union Frag { v16bf v; uint4 q[2]; };

// Tensor Data Mover availability (guarded so compile can never break)
#if defined(__has_builtin)
#if __has_builtin(__builtin_amdgcn_tensor_load_to_lds) && \
    __has_builtin(__builtin_amdgcn_s_wait_tensorcnt)
#define HAVE_TDM 1
#endif
#endif
#ifndef HAVE_TDM
#define HAVE_TDM 0
#endif

// fp32 -> bf16 (round-to-nearest-even), stored as ushort in buffers
__device__ __forceinline__ unsigned short f2bf(float f) {
  unsigned int u = __builtin_bit_cast(unsigned int, f);
  unsigned int lsb = (u >> 16) & 1u;
  u += 0x7fffu + lsb;
  return (unsigned short)(u >> 16);
}

// ---------------------------------------------------------------------------
// Weight pack: OIHW fp32 -> bf16 [k/32][Cout][32], k = ci*KH*KW + kh*KW + kw
// (source flat index is exactly co*Ktot + k)
// ---------------------------------------------------------------------------
__global__ void k_pack_w(const float* __restrict__ w, unsigned short* __restrict__ wp,
                         int Cout, int Ktot) {
  int idx = blockIdx.x * 256 + threadIdx.x;
  if (idx >= Cout * Ktot) return;
  int co = idx / Ktot, k = idx - co * Ktot;
  wp[(size_t)(k >> 5) * Cout * 32 + (size_t)co * 32 + (k & 31)] = f2bf(w[idx]);
}

// BN fold: scale = g*rsqrt(v+eps); shift = (convBias - mean)*scale + beta
__global__ void k_fold_bn(const float* __restrict__ g, const float* __restrict__ beta,
                          const float* __restrict__ mean, const float* __restrict__ var,
                          const float* __restrict__ convb,
                          float* __restrict__ scale, float* __restrict__ shift, int C) {
  int c = blockIdx.x * 256 + threadIdx.x;
  if (c >= C) return;
  float s = g[c] * rsqrtf(var[c] + 1e-5f);
  scale[c] = s;
  shift[c] = (convb[c] - mean[c]) * s + beta[c];
}

// channel concat (B,C,HW)+(B,C,HW) -> bf16 (B,2C,HW)
__global__ void k_concat_h(const float* __restrict__ a, const float* __restrict__ b,
                           unsigned short* __restrict__ out, int B, int C, int HW) {
  int idx = blockIdx.x * 256 + threadIdx.x;
  int tot = B * 2 * C * HW;
  if (idx >= tot) return;
  int hw = idx % HW;
  int c  = (idx / HW) % (2 * C);
  int bb = idx / (HW * 2 * C);
  float v = (c < C) ? a[(size_t)(bb * C + c) * HW + hw]
                    : b[(size_t)(bb * C + (c - C)) * HW + hw];
  out[idx] = f2bf(v);
}

// 2x bilinear upsample (align_corners=False) * 2.0 -> bf16
__global__ void k_up2x(const float* __restrict__ in, unsigned short* __restrict__ out,
                       int NC, int H, int W) {
  int Ho = 2 * H, Wo = 2 * W;
  int tot = NC * Ho * Wo;
  int idx = blockIdx.x * 256 + threadIdx.x;
  if (idx >= tot) return;
  int wo = idx % Wo;
  int ho = (idx / Wo) % Ho;
  int nc = idx / (Wo * Ho);
  float sy = (ho + 0.5f) * 0.5f - 0.5f;
  float sx = (wo + 0.5f) * 0.5f - 0.5f;
  int y0 = (int)floorf(sy); float fy = sy - (float)y0;
  int x0 = (int)floorf(sx); float fx = sx - (float)x0;
  int y1 = y0 + 1, x1 = x0 + 1;
  y0 = y0 < 0 ? 0 : (y0 > H - 1 ? H - 1 : y0);
  y1 = y1 < 0 ? 0 : (y1 > H - 1 ? H - 1 : y1);
  x0 = x0 < 0 ? 0 : (x0 > W - 1 ? W - 1 : x0);
  x1 = x1 < 0 ? 0 : (x1 > W - 1 ? W - 1 : x1);
  const float* p = in + (size_t)nc * H * W;
  float v = (1.f - fy) * ((1.f - fx) * p[y0 * W + x0] + fx * p[y0 * W + x1]) +
            fy * ((1.f - fx) * p[y1 * W + x0] + fx * p[y1 * W + x1]);
  out[idx] = f2bf(v * 2.0f);
}

// global average pool: one block per (b,c)
__global__ void __launch_bounds__(256) k_gap(const float* __restrict__ in,
                                             float* __restrict__ out, int HW) {
  __shared__ float red[256];
  int bc = blockIdx.x;
  const float* p = in + (size_t)bc * HW;
  float s = 0.f;
  for (int i = threadIdx.x; i < HW; i += 256) s += p[i];
  red[threadIdx.x] = s;
  __syncthreads();
  for (int st = 128; st > 0; st >>= 1) {
    if ((int)threadIdx.x < st) red[threadIdx.x] += red[threadIdx.x + st];
    __syncthreads();
  }
  if (threadIdx.x == 0) out[bc] = red[0] / (float)HW;
}

// tiny fully-connected: out[b][o] = act(dot(in[b], W[o]) + bias[o])
// act: 0=linear, 1=lrelu(slope), 2=relu
__global__ void k_fc(const float* __restrict__ in, const float* __restrict__ w,
                     const float* __restrict__ bias, float* __restrict__ out,
                     int IN, int OUT, int act, float slope) {
  int o = blockIdx.y * 256 + threadIdx.x;
  int b = blockIdx.x;
  if (o >= OUT) return;
  const float* x  = in + (size_t)b * IN;
  const float* wr = w + (size_t)o * IN;
  float s = bias ? bias[o] : 0.f;
  for (int i = 0; i < IN; ++i) s += x[i] * wr[i];
  if (act == 1) s = s >= 0.f ? s : s * slope;
  else if (act == 2) s = fmaxf(s, 0.f);
  out[(size_t)b * OUT + o] = s;
}

// dynamic depthwise 3x3, pad 1, per-(b,c) kernel from kvec (B, C*9), lrelu 0.1 -> bf16
__global__ void k_dw3(const float* __restrict__ x, const float* __restrict__ kvec,
                      unsigned short* __restrict__ out, int B, int C, int H, int W) {
  int idx = blockIdx.x * 256 + threadIdx.x;
  int tot = B * C * H * W;
  if (idx >= tot) return;
  int w0 = idx % W;
  int h0 = (idx / W) % H;
  int c  = (idx / (W * H)) % C;
  int b  = idx / (W * H * C);
  const float* kp = kvec + (size_t)b * C * 9 + (size_t)c * 9;
  const float* p  = x + (size_t)(b * C + c) * H * W;
  float s = 0.f;
#pragma unroll
  for (int kh = 0; kh < 3; ++kh) {
    int ih = h0 + kh - 1;
    if (ih < 0 || ih >= H) continue;
#pragma unroll
    for (int kw = 0; kw < 3; ++kw) {
      int iw = w0 + kw - 1;
      if (iw < 0 || iw >= W) continue;
      s += p[ih * W + iw] * kp[kh * 3 + kw];
    }
  }
  s = s >= 0.f ? s : 0.1f * s;
  out[idx] = f2bf(s);
}

// ---------------------------------------------------------------------------
// Implicit-GEMM conv via v_wmma_f32_16x16x32_bf16.
// GEMM: M = B*Ho*Wo (b,oh,ow), N = Cout, K = Cin*KHW*KHW (ci,kh,kw).
// Block: 256 threads = 8 waves; 64(M) x 64(N) tile; K-step 32, double-buffered.
// B (weight) tile is a contiguous 4KB slice of the packed weight buffer ->
// staged by the Tensor Data Mover (tensor_load_to_lds, TENSORcnt), with the
// descriptor's pad feature (16B every 64B) creating the 40-halfword LDS row
// stride that de-conflicts banks. A tile (im2col gather) is software-pipelined
// through registers so global latency overlaps the WMMAs.
// Requires Ktot % 32 == 0 and Cout % 64 == 0 (true for every conv here).
// ---------------------------------------------------------------------------
__global__ void __launch_bounds__(256) k_conv_wmma(
    const unsigned short* __restrict__ xh,   // bf16 activations NCHW
    const unsigned short* __restrict__ wp,   // packed bf16 weights [k/32][Cout][32]
    int Cin, int H, int W, int Batch,
    int Cout, int Ho, int Wo,
    int stride, int pad, int KHW,
    const float* __restrict__ scale,  // per-Cout scale (nullptr -> 1)
    const float* __restrict__ shift,  // per-Cout shift/bias (nullptr -> 0)
    const float* __restrict__ resid,  // residual add, NCHW out dims (nullable)
    const float* __restrict__ att,    // (B,Cout) attention (nullable)
    const float* __restrict__ xatt,   // x for x*att, NCHW out dims (nullable)
    float slope, int act,             // act: 0=none, 1=lrelu(slope)
    float* __restrict__ outF,         // fp32 out (nullable)
    unsigned short* __restrict__ outH)// bf16 out (nullable)
{
  __shared__ __align__(16) unsigned short As[2][64][40];
  __shared__ __align__(16) unsigned short Bs[2][64][40];

  const int tid  = threadIdx.x;
  const int wave = tid >> 5;
  const int lane = tid & 31;
  const int mTile = blockIdx.x;
  const int nTile = blockIdx.y;
  const int HoWo = Ho * Wo;
  const int Mtot = Batch * HoWo;
  const int KK   = KHW * KHW;
  const int Ktot = Cin * KK;

  const int ms  = wave & 3;          // M subtile 0..3
  const int ns0 = (wave >> 2) << 1;  // N subtile pair base (0 or 2)

  v8f acc0 = {};
  v8f acc1 = {};

  // staging assignments
  const int a_k  = tid >> 3;         // 0..31 (K within tile)
  const int a_m0 = (tid & 7) << 3;   // 0..56 (M base, 8 rows each)
#if !HAVE_TDM
  const int b_co = tid >> 2;         // 0..63 (N within tile)
  const int b_k0 = (tid & 3) << 3;   // 0,8,16,24
#endif

  // im2col gather for K-step k0 into registers (8 values, fixed k per thread)
  auto gatherA = [&](int k0, unsigned short* av) {
    int k = k0 + a_k;
    int ci = k / KK, r = k - ci * KK;
    int kh = r / KHW, kw = r - kh * KHW;
#pragma unroll
    for (int j = 0; j < 8; ++j) {
      int m = mTile * 64 + a_m0 + j;
      unsigned short v = 0;
      if (m < Mtot) {
        int b   = m / HoWo;
        int rem = m - b * HoWo;
        int oh  = rem / Wo, ow = rem - oh * Wo;
        int ih  = oh * stride - pad + kh;
        int iw  = ow * stride - pad + kw;
        if (ih >= 0 && ih < H && iw >= 0 && iw < W)
          v = xh[((size_t)(b * Cin + ci) * H + ih) * W + iw];
      }
      av[j] = v;
    }
  };
  auto storeA = [&](int buf, const unsigned short* av) {
#pragma unroll
    for (int j = 0; j < 8; ++j) As[buf][a_m0 + j][a_k] = av[j];
  };

#if HAVE_TDM
  // Tensor Data Mover: contiguous 64x64B weight slice -> LDS, +16B pad per 64B.
  auto stageB = [&](int buf, int k0) {
    const unsigned short* src = wp + (size_t)(k0 >> 5) * Cout * 32 +
                                (size_t)nTile * 64 * 32;
    unsigned long long ga = (unsigned long long)(size_t)src;
    unsigned lds = (unsigned)(size_t)&Bs[buf][0][0];
    u32x4 g0 = {1u,                                     // count=1 (valid D#)
                lds,                                    // lds_addr
                (unsigned)ga,                           // global_addr[31:0]
                (unsigned)((ga >> 32) & 0x01ffffffu) | 0x80000000u}; // [56:32]|type=2
    i32x8 g1 = {(int)((1u << 16) |      // data_size = 2 bytes
                      (1u << 20) |      // pad_enable
                      (3u << 22) |      // pad_interval: every 16 DWORDs (64B)
                      (3u << 25)),      // pad_amount: 4 DWORDs (16B)
                (int)(2048u << 16),     // tensor_dim0 = 2048 elems (low 16)
                (int)(1u << 16),        // tensor_dim0 hi=0; tensor_dim1 = 1
                (int)(2048u << 16),     // tensor_dim1 hi=0; tile_dim0 = 2048
                1,                      // tile_dim1 = 1; tile_dim2 = 0
                2048,                   // tensor_dim0_stride (low 32)
                0, 0};
    i32x4 z4 = {0, 0, 0, 0};
#if __clang_major__ >= 23
    i32x8 z8 = {0, 0, 0, 0, 0, 0, 0, 0};
    __builtin_amdgcn_tensor_load_to_lds(g0, g1, z4, z4, z8, 0);
#else
    __builtin_amdgcn_tensor_load_to_lds(g0, g1, z4, z4, 0);
#endif
  };
#else
  auto stageB = [&](int buf, int k0) {
    const unsigned short* src = wp + (size_t)(k0 >> 5) * Cout * 32 +
                                (size_t)(nTile * 64 + b_co) * 32 + b_k0;
    *(uint4*)&Bs[buf][b_co][b_k0] = *(const uint4*)src;
  };
#endif

  // ---- prologue: stage buffer 0 ----
#if HAVE_TDM
  if (wave == 0) stageB(0, 0);
#else
  stageB(0, 0);
#endif
  {
    unsigned short av[8];
    gatherA(0, av);
    storeA(0, av);
  }
#if HAVE_TDM
  if (wave == 0) __builtin_amdgcn_s_wait_tensorcnt(0);
#endif
  __syncthreads();

  // ---- pipelined K loop ----
  for (int k0 = 0; k0 < Ktot; k0 += 32) {
    const int cur  = (k0 >> 5) & 1;
    const int k1   = k0 + 32;
    const bool more = (k1 < Ktot);

    unsigned short av[8];
    if (more) {
#if HAVE_TDM
      if (wave == 0) stageB(cur ^ 1, k1);   // DMA overlaps the WMMAs below
#else
      stageB(cur ^ 1, k1);
#endif
      gatherA(k1, av);                      // global loads overlap WMMAs
    }

    // fragments from LDS (16B ds_load chunks per ISA 16-bit A/B lane layouts)
    Frag fa, fb0, fb1;
    {
      int ml    = ms * 16 + (lane & 15);
      int khalf = (lane >> 4) << 3;  // 0 or 8
      fa.q[0] = *(const uint4*)&As[cur][ml][khalf];
      fa.q[1] = *(const uint4*)&As[cur][ml][16 + khalf];
      int nl = ns0 * 16 + (lane & 15);
      int lh = (lane >> 4) << 3;     // 0 or 8
      const uint4* pb0 = (const uint4*)&Bs[cur][nl][lh];
      fb0.q[0] = pb0[0]; fb0.q[1] = pb0[2];
      const uint4* pb1 = (const uint4*)&Bs[cur][nl + 16][lh];
      fb1.q[0] = pb1[0]; fb1.q[1] = pb1[2];
    }
    acc0 = __builtin_amdgcn_wmma_f32_16x16x32_bf16(false, fa.v, false, fb0.v,
                                                   (short)0, acc0, false, false);
    acc1 = __builtin_amdgcn_wmma_f32_16x16x32_bf16(false, fa.v, false, fb1.v,
                                                   (short)0, acc1, false, false);

    if (more) storeA(cur ^ 1, av);
#if HAVE_TDM
    if (more && wave == 0) __builtin_amdgcn_s_wait_tensorcnt(0);
#endif
    __syncthreads();
  }

  // ---- epilogue ----
  // C/D layout: lanes 0-15: N=lane, VGPR r -> M=r; lanes 16-31: N=lane-16, M=8+r
  const int n_local = lane & 15;
  const int m_hi    = (lane >> 4) << 3;
#pragma unroll
  for (int sub = 0; sub < 2; ++sub) {
    v8f accv;
    if (sub) accv = acc1; else accv = acc0;
    int co = nTile * 64 + (ns0 + sub) * 16 + n_local;
    float sc = scale ? scale[co] : 1.0f;
    float sh = shift ? shift[co] : 0.0f;
#pragma unroll
    for (int r = 0; r < 8; ++r) {
      int m = mTile * 64 + ms * 16 + m_hi + r;
      if (m >= Mtot) continue;
      int b   = m / HoWo;
      int rem = m - b * HoWo;
      size_t oidx = (size_t)(b * Cout + co) * HoWo + rem;
      float z = accv[r] * sc + sh;
      if (att)   z += xatt[oidx] * att[b * Cout + co];
      if (resid) z += resid[oidx];
      if (act == 1) z = z >= 0.f ? z : z * slope;
      if (outF) outF[oidx] = z;
      if (outH) outH[oidx] = f2bf(z);
    }
  }
}

// ---------------------------------------------------------------------------
// Host orchestration
// ---------------------------------------------------------------------------
extern "C" void kernel_launch(void* const* d_in, const int* in_sizes, int n_in,
                              void* d_out, int out_size, void* d_ws, size_t ws_size,
                              hipStream_t stream) {
  (void)in_sizes; (void)n_in; (void)out_size; (void)ws_size;

  auto in = [&](int i) -> const float* { return (const float*)d_in[i]; };

  // Input indices per setup_inputs() dict order:
  // 0 mv, 1 mv_aux2, 2 mv_aux3, 3 aw, 4 ab, 5 c1w, 6 c1b, 7 c2w, 8 c2b,
  // then per prefix (d1_ at 9, d2_ at 37), offsets:
  //  +0 kw1 +1 kw2 +2 cw +3 cb +4 ca1 +5 ca2 +6 e1w +7 e1b +8 e2w +9 e2b
  //  +10 e3w +11 e3b +12..15 bn1{g,b,m,v} +16..19 bn2{g,b,m,v}
  //  +20..23 bn3{g,b,m,v} +24 m1w +25 m1b +26 m2w +27 m2b
  const float* mv   = in(0);
  const float* aux2 = in(1);
  const float* aux3 = in(2);

  // Workspace layout (~138 MB)
  char* wsb = (char*)d_ws;
  size_t off = 0;
  auto alloc = [&](size_t bytes) -> void* {
    void* p = wsb + off;
    off = (off + bytes + 255) & ~(size_t)255;
    return p;
  };
  unsigned short* concat_h = (unsigned short*)alloc(8388608ull * 2);   // (4,128,128,128) bf16
  float*          mvaux_f  = (float*)alloc(4194304ull * 4);            // (4,64,128,128)
  unsigned short* yup_h    = (unsigned short*)alloc(16777216ull * 2);  // (4,64,256,256) bf16
  unsigned short* enc1_h   = (unsigned short*)alloc(8388608ull * 2);   // (4,128,128,128) bf16
  unsigned short* enc2_h   = (unsigned short*)alloc(2032128ull * 2);   // (4,128,63,63) bf16
  float*          enc3_f   = (float*)alloc(3810304ull * 4);            // (4,256,61,61)
  unsigned short* dw_h     = (unsigned short*)alloc(4194304ull * 2);   // (4,64,128,128) bf16
  unsigned short* x1_h     = (unsigned short*)alloc(4194304ull * 2);   // (4,64,128,128) bf16
  float*          y2_f     = (float*)alloc(4194304ull * 4);            // (4,64,128,128)
  unsigned short* wp       = (unsigned short*)alloc(294912ull * 2);    // max packed weights
  float* scaleB = (float*)alloc(256 * 4);
  float* shiftB = (float*)alloc(256 * 4);
  float* fea    = (float*)alloc(1024 * 4);
  float* hbuf   = (float*)alloc(1024 * 4);
  float* rep    = (float*)alloc(1024 * 4);
  float* t128   = (float*)alloc(512 * 4);
  float* kvec   = (float*)alloc(2304 * 4);
  float* attb   = (float*)alloc(256 * 4);
  float* t32    = (float*)alloc(128 * 4);

  // Generic conv launcher (pack + wmma conv). Serial on one stream -> wp,
  // scaleB/shiftB can be reused across convs.
  auto conv = [&](const unsigned short* xh, const float* wsrc,
                  int Cin, int H, int W, int Cout, int strideC, int pad, int KHW,
                  const float* scale, const float* shift,
                  const float* resid, const float* attp, const float* xattp,
                  float slope, int act, float* outF, unsigned short* outHp) {
    int Ktot = Cin * KHW * KHW;
    int nw = Cout * Ktot;
    k_pack_w<<<dim3((nw + 255) / 256), dim3(256), 0, stream>>>(wsrc, wp, Cout, Ktot);
    int Ho = (H + 2 * pad - KHW) / strideC + 1;
    int Wo = (W + 2 * pad - KHW) / strideC + 1;
    int Mtot = 4 * Ho * Wo;
    dim3 grid((Mtot + 63) / 64, Cout / 64);
    k_conv_wmma<<<grid, dim3(256), 0, stream>>>(
        xh, wp, Cin, H, W, 4, Cout, Ho, Wo, strideC, pad, KHW,
        scale, shift, resid, attp, xattp, slope, act, outF, outHp);
  };

  // da_conv block: x,y are fp32 (4,64,128,128); result (after outer lrelu 0.1)
  // written as bf16 to outh.
  auto da = [&](int db, const float* x, const float* y, unsigned short* outh) {
    // y_up = bilinear2x(y) * 2 -> bf16
    k_up2x<<<dim3(16777216 / 256), dim3(256), 0, stream>>>(y, yup_h, 4 * 64, 128, 128);
    // encoder: e1 (64->128, s2 p1), e2 (128->128, s2 p0), e3 (128->256, s1 p0)
    k_fold_bn<<<dim3(1), dim3(256), 0, stream>>>(in(db+12), in(db+13), in(db+14), in(db+15),
                                                 in(db+7), scaleB, shiftB, 128);
    conv(yup_h, in(db+6), 64, 256, 256, 128, 2, 1, 3,
         scaleB, shiftB, nullptr, nullptr, nullptr, 0.1f, 1, nullptr, enc1_h);
    k_fold_bn<<<dim3(1), dim3(256), 0, stream>>>(in(db+16), in(db+17), in(db+18), in(db+19),
                                                 in(db+9), scaleB, shiftB, 128);
    conv(enc1_h, in(db+8), 128, 128, 128, 128, 2, 0, 3,
         scaleB, shiftB, nullptr, nullptr, nullptr, 0.1f, 1, nullptr, enc2_h);
    k_fold_bn<<<dim3(1), dim3(256), 0, stream>>>(in(db+20), in(db+21), in(db+22), in(db+23),
                                                 in(db+11), scaleB, shiftB, 256);
    conv(enc2_h, in(db+10), 128, 63, 63, 256, 1, 0, 3,
         scaleB, shiftB, nullptr, nullptr, nullptr, 0.1f, 1, enc3_f, nullptr);
    // GAP + MLPs
    k_gap<<<dim3(4 * 256), dim3(256), 0, stream>>>(enc3_f, fea, 61 * 61);
    k_fc<<<dim3(4, 1), dim3(256), 0, stream>>>(fea, in(db+24), in(db+25), hbuf, 256, 256, 1, 0.1f);
    k_fc<<<dim3(4, 1), dim3(256), 0, stream>>>(hbuf, in(db+26), in(db+27), rep, 256, 256, 0, 0.f);
    // dynamic kernels
    k_fc<<<dim3(4, 1), dim3(256), 0, stream>>>(rep, in(db+0), nullptr, t128, 256, 128, 1, 0.1f);
    k_fc<<<dim3(4, 3), dim3(256), 0, stream>>>(t128, in(db+1), nullptr, kvec, 128, 576, 0, 0.f);
    // channel attention
    k_fc<<<dim3(4, 1), dim3(256), 0, stream>>>(rep, in(db+4), nullptr, t32, 256, 32, 1, 0.1f);
    k_fc<<<dim3(4, 1), dim3(256), 0, stream>>>(t32, in(db+5), nullptr, attb, 32, 64, 2, 0.f);
    // depthwise dynamic conv + lrelu 0.1 -> bf16
    k_dw3<<<dim3(4194304 / 256), dim3(256), 0, stream>>>(x, kvec, dw_h, 4, 64, 128, 128);
    // 1x1 conv cw + cb, + x*att, outer lrelu 0.1 -> bf16
    conv(dw_h, in(db+2), 64, 128, 128, 64, 1, 0, 1,
         nullptr, in(db+3), nullptr, attb, x, 0.1f, 1, nullptr, outh);
  };

  // ---- forward ----
  // mv_aux = lrelu_0.01(conv(concat(aux2,aux3), aw) + ab)
  k_concat_h<<<dim3(8388608 / 256), dim3(256), 0, stream>>>(aux2, aux3, concat_h, 4, 64, 16384);
  conv(concat_h, in(3), 128, 128, 128, 64, 1, 1, 3,
       nullptr, in(4), nullptr, nullptr, nullptr, 0.01f, 1, mvaux_f, nullptr);
  // d1: da_conv(x=mv_aux, y=mv_aux), outer lrelu 0.1 folded
  da(9, mvaux_f, mvaux_f, x1_h);
  // c1: conv + bias + lrelu 0.1 -> fp32 (feeds d2's upsample)
  conv(x1_h, in(5), 64, 128, 128, 64, 1, 1, 3,
       nullptr, in(6), nullptr, nullptr, nullptr, 0.1f, 1, y2_f, nullptr);
  // d2: da_conv(x=mv, y=c1-out), outer lrelu 0.1 folded (reuse x1_h)
  da(37, mv, y2_f, x1_h);
  // c2: conv + bias + mv residual -> d_out (fp32)
  conv(x1_h, in(7), 64, 128, 128, 64, 1, 1, 3,
       nullptr, in(8), mv, nullptr, nullptr, 0.f, 0, (float*)d_out, nullptr);
}
